// CausalSelfAttention_32521492366045
// MI455X (gfx1250) — matrix operand order repaired and checked by reading, not compiled, not run on hardware
//
#include <hip/hip_runtime.h>

// ---------------------------------------------------------------------------
// CausalSelfAttention forward for MI455X (gfx1250), wave32 + WMMA bf16 + TDM.
// B=2, T=2048, C=1024, H=16, D=64
// ---------------------------------------------------------------------------

typedef __bf16 bf16;
typedef bf16            v16bf  __attribute__((ext_vector_type(16)));
typedef float           v8f    __attribute__((ext_vector_type(8)));
typedef unsigned int    u32x4  __attribute__((ext_vector_type(4)));
typedef int             i32x4  __attribute__((ext_vector_type(4)));
typedef int             i32x8  __attribute__((ext_vector_type(8)));
typedef unsigned short  u16;
typedef unsigned int    u32;
typedef unsigned long long u64;

#define B_  2
#define T_  2048
#define C_  1024
#define H_  16
#define D_  64
#define LD3C 3072   // 3*C

union FragB16 {            // 16 bf16 = 8 VGPRs: one WMMA A or B operand
  v16bf v;
  u32x4 q[2];
  u16   h[16];
};

__device__ __forceinline__ u16 f32_to_bf16(float f) {
  u32 u = __builtin_bit_cast(u32, f);
  u32 r = 0x7FFFu + ((u >> 16) & 1u);   // round-to-nearest-even
  u += r;
  return (u16)(u >> 16);
}

// Low 32 bits of a flat pointer into LDS == byte offset within the
// workgroup's LDS allocation (aperture lives in addr[63:32]).
__device__ __forceinline__ u32 lds_offset(const void* p) {
  return (u32)(unsigned long)p;
}

// ---------------------------------------------------------------------------
// TDM: issue a 2D tile load Global -> LDS.
// Tile: tile_d1 rows x tile_d0 contiguous elements (bf16), row stride
// `stride0` elements. D# built per CDNA5 ISA 8.3/8.4 bitfields.
// clang-23 toolchain: 6-arg builtin (g0, g1, g2, g3, g4, cpol).
// ---------------------------------------------------------------------------
__device__ __forceinline__ void tdm_load_2d(u32 lds_addr, const u16* gptr,
                                            int tile_d0, int tile_d1,
                                            long stride0) {
  const u64 ga = (u64)gptr;
  const u64 s0 = (u64)stride0;
  u32x4 g0;
  g0[0] = 1u;                                          // count=1, user desc
  g0[1] = lds_addr;                                    // lds_addr [63:32]
  g0[2] = (u32)ga;                                     // global_addr lo
  g0[3] = (u32)((ga >> 32) & 0x01FFFFFFu) | 0x80000000u; // addr hi | type=2
  i32x8 g1;
  g1[0] = 1 << 16;                                     // data_size=2 bytes
  g1[1] = (tile_d0 & 0xFFFF) << 16;                    // tensor_dim0 lo16
  g1[2] = (tile_d0 >> 16) | ((tile_d1 & 0xFFFF) << 16);// t_d0 hi | t_d1 lo
  g1[3] = (tile_d1 >> 16) | (tile_d0 << 16);           // t_d1 hi | tile_dim0
  g1[4] = tile_d1;                                     // tile_dim1 (dim2=0)
  g1[5] = (int)(s0 & 0xFFFFFFFFu);                     // dim0_stride lo32
  g1[6] = (int)((s0 >> 32) & 0xFFFFu);                 // dim0_stride hi16
  g1[7] = 0;
  i32x4 g2 = {0, 0, 0, 0};                             // 2D: group2 unused
  i32x4 g3 = {0, 0, 0, 0};                             // 2D: group3 unused
  i32x8 g4 = {0, 0, 0, 0, 0, 0, 0, 0};                 // trailing group: zero
  __builtin_amdgcn_tensor_load_to_lds(g0, g1, g2, g3, g4, 0);
}

// ---------------------------------------------------------------------------
// Kernel 1: elementwise fp32 -> bf16 (grid-stride)
// ---------------------------------------------------------------------------
__global__ void __launch_bounds__(256)
cvt_f32_bf16_kernel(const float* __restrict__ in, u16* __restrict__ out, int n) {
  int i = blockIdx.x * blockDim.x + threadIdx.x;
  int stride = gridDim.x * blockDim.x;
  for (; i < n; i += stride) out[i] = f32_to_bf16(in[i]);
}

// ---------------------------------------------------------------------------
// Kernel 2/4: C[M,N] = A[M,K] @ B[K,N], bf16 in, fp32 accumulate.
// Workgroup (8 waves) owns a 128x128 tile; K stepped by 32 with TDM
// double-buffered LDS staging. Wave w = (wm = w>>1 in 0..3, wn = w&1):
// computes 32 rows x 64 cols = 2x4 WMMA tiles, A-frags reused 4x.
// ---------------------------------------------------------------------------
template <bool OUT_BF16>
__global__ void __launch_bounds__(256)
gemm_bf16_wmma(const u16* __restrict__ A, const u16* __restrict__ Bm,
               void* __restrict__ Cm, int M, int N, int K) {
  __shared__ __align__(16) u16 As[2][128 * 32];   // [row][k]  8KB x2
  __shared__ __align__(16) u16 Bs[2][32 * 128];   // [k][col]  8KB x2

  const int lane = threadIdx.x & 31;
  const int wid  = threadIdx.x >> 5;
  const int ntn  = N >> 7;
  const int bm   = blockIdx.x / ntn;
  const int bn   = blockIdx.x % ntn;
  const int wm   = wid >> 1;          // 0..3  -> rows wm*32
  const int wn   = wid & 1;           // 0..1  -> cols wn*64
  const int hs   = lane >> 4;
  const int l16  = lane & 15;

  const u16* Ablk = A  + (size_t)(bm * 128) * K;
  const u16* Bblk = Bm + bn * 128;
  const int  nk   = K >> 5;

  // prologue: DMA first K-slab into buffer 0 (wave 0 only; TENSORcnt is
  // per-wave, so the issuing wave waits, then the barrier publishes LDS)
  if (threadIdx.x < 32) {
    tdm_load_2d(lds_offset(&As[0][0]), Ablk, 32, 128, K);
    tdm_load_2d(lds_offset(&Bs[0][0]), Bblk, 128, 32, N);
  }

  v8f acc[2][4] = {};

  for (int kt = 0; kt < nk; ++kt) {
    const int buf = kt & 1;
    if (threadIdx.x < 32) {
      if (kt + 1 < nk) {
        const int nb = buf ^ 1;
        tdm_load_2d(lds_offset(&As[nb][0]), Ablk + (kt + 1) * 32, 32, 128, K);
        tdm_load_2d(lds_offset(&Bs[nb][0]),
                    Bblk + (size_t)(kt + 1) * 32 * N, 128, 32, N);
        __builtin_amdgcn_s_wait_tensorcnt(2);   // current buf's 2 DMAs done
      } else {
        __builtin_amdgcn_s_wait_tensorcnt(0);
      }
    }
    __syncthreads();                            // buf is ready for everyone

    const u16* as = &As[buf][0];
    const u16* bs = &Bs[buf][0];

    // A-frags: rows wm*32 + l16 and +16 (A layout: lane row = lane&15,
    // elems 0..7 -> k=8*hs..+7, elems 8..15 -> k=16+8*hs..+7)
    FragB16 a0, a1;
    {
      const u16* r0 = as + (wm * 32 + l16) * 32;
      a0.q[0] = *(const u32x4*)(r0 + 8 * hs);
      a0.q[1] = *(const u32x4*)(r0 + 16 + 8 * hs);
      const u16* r1 = r0 + 16 * 32;
      a1.q[0] = *(const u32x4*)(r1 + 8 * hs);
      a1.q[1] = *(const u32x4*)(r1 + 16 + 8 * hs);
    }
    // B-frags: column n = lane&15, elems -> k in A's table order
    const u16* brow = bs + wn * 64 + l16;
#pragma unroll
    for (int j = 0; j < 4; ++j) {
      FragB16 b;
#pragma unroll
      for (int s = 0; s < 16; ++s) {
        const int kk = (s < 8) ? (8 * hs + s) : (16 + 8 * hs + (s - 8));
        b.h[s] = brow[kk * 128 + j * 16];
      }
      acc[0][j] = __builtin_amdgcn_wmma_f32_16x16x32_bf16(
          false, a0.v, false, b.v, (short)0, acc[0][j], false, false);
      acc[1][j] = __builtin_amdgcn_wmma_f32_16x16x32_bf16(
          false, a1.v, false, b.v, (short)0, acc[1][j], false, false);
    }
    __syncthreads();                            // done reading buf
  }

  // C/D layout: VGPR r holds row m = r + 8*hs, lane column n = lane&15
#pragma unroll
  for (int mi = 0; mi < 2; ++mi) {
#pragma unroll
    for (int r = 0; r < 8; ++r) {
      const int m = bm * 128 + wm * 32 + mi * 16 + r + 8 * hs;
      const size_t row = (size_t)m * N + bn * 128 + wn * 64 + l16;
#pragma unroll
      for (int j = 0; j < 4; ++j) {
        if (OUT_BF16) ((u16*)Cm)[row + j * 16]  = f32_to_bf16(acc[mi][j][r]);
        else          ((float*)Cm)[row + j * 16] = acc[mi][j][r];
      }
    }
  }
}

// ---------------------------------------------------------------------------
// Kernel 3: flash attention. One wave per (b, h, 16-query tile).
// S^T = K @ Q^T  (softmax rows = queries land in lanes), online softmax,
// P re-layout via per-wave LDS tile, O = alpha*O + P @ V via WMMA.
// ---------------------------------------------------------------------------
__global__ void __launch_bounds__(128)
attn_flash_kernel(const u16* __restrict__ qkv, u16* __restrict__ y) {
  __shared__ __align__(16) u16 ldsP[4][16 * 32];   // per-wave 16x32 bf16 P tile

  const int lane = threadIdx.x & 31;
  const int wid  = threadIdx.x >> 5;
  const int job  = blockIdx.x * 4 + wid;           // 4096 jobs total
  const int qi   = job & 127;                      // query tile (T/16 = 128)
  const int h    = (job >> 7) & (H_ - 1);
  const int b    = job >> 11;

  const int hs  = lane >> 4;
  const int l16 = lane & 15;

  const u16* base = qkv + (size_t)b * T_ * LD3C + h * D_;

  // Q^T B-frags, loaded once: column n = query lane&15, elems -> d in table order.
  FragB16 Qt0, Qt1;
  {
    const u16* qrow = base + (size_t)(qi * 16 + l16) * LD3C;
    Qt0.q[0] = *(const u32x4*)(qrow +      8 * hs);
    Qt0.q[1] = *(const u32x4*)(qrow + 16 + 8 * hs);
    Qt1.q[0] = *(const u32x4*)(qrow + 32 + 8 * hs);
    Qt1.q[1] = *(const u32x4*)(qrow + 48 + 8 * hs);
  }

  v8f O0 = {}, O1 = {}, O2 = {}, O3 = {};      // 16 queries x 64 dims, C-layout
  float m_run = -__builtin_inff();
  float l_run = 0.f;
  const int qg_lane = qi * 16 + l16;           // this lane's softmax row
  const int kbmax   = (qi * 16 + 15) >> 5;     // inclusive 32-key blocks
  u16* myP = &ldsP[wid][0];

  for (int kb = 0; kb <= kbmax; ++kb) {
    // prefetch next block's K and V rows (one row per lane)
    if (kb < kbmax) {
      const size_t nrow = (size_t)(kb * 32 + 32 + lane) * LD3C;
      __builtin_prefetch(base + nrow + C_,     0, 0);
      __builtin_prefetch(base + nrow + 2 * C_, 0, 0);
    }

    // ---- S^T = K_block @ Q^T, two 16-key subtiles, accumulate over d=64 ----
    v8f S[2];
#pragma unroll
    for (int s = 0; s < 2; ++s) {
      int keyrow = kb * 32 + s * 16 + l16;
      if (keyrow > T_ - 1) keyrow = T_ - 1;    // clamped rows masked below
      const u16* kbase = base + (size_t)keyrow * LD3C + C_;
      FragB16 k0, k1;
      k0.q[0] = *(const u32x4*)(kbase +      8 * hs);
      k0.q[1] = *(const u32x4*)(kbase + 16 + 8 * hs);
      k1.q[0] = *(const u32x4*)(kbase + 32 + 8 * hs);
      k1.q[1] = *(const u32x4*)(kbase + 48 + 8 * hs);
      v8f sa = {};
      sa = __builtin_amdgcn_wmma_f32_16x16x32_bf16(false, k0.v, false, Qt0.v, (short)0, sa, false, false);
      sa = __builtin_amdgcn_wmma_f32_16x16x32_bf16(false, k1.v, false, Qt1.v, (short)0, sa, false, false);
      S[s] = sa;
    }

    // ---- scale 1/sqrt(64), causal mask, block row-max ----
    const bool need_mask = (kb * 32 + 31) > (qi * 16);
    float blockmax = -__builtin_inff();
#pragma unroll
    for (int s = 0; s < 2; ++s) {
#pragma unroll
      for (int r = 0; r < 8; ++r) {
        float v = S[s][r] * 0.125f;
        const int key_g = kb * 32 + s * 16 + r + 8 * hs;
        if (need_mask && key_g > qg_lane) v = -__builtin_inff();
        S[s][r] = v;
        blockmax = fmaxf(blockmax, v);
      }
    }
    blockmax = fmaxf(blockmax, __shfl_xor(blockmax, 16, 32));
    const float m_new = fmaxf(m_run, blockmax);
    const float alpha = __expf(m_run - m_new);

    // ---- P = exp(S - m), row sums; stage P into LDS [query][key] ----
    float rsum = 0.f;
#pragma unroll
    for (int s = 0; s < 2; ++s) {
#pragma unroll
      for (int r = 0; r < 8; ++r) {
        const float p = __expf(S[s][r] - m_new);
        rsum += p;
        const int key_l = s * 16 + r + 8 * hs;
        myP[l16 * 32 + key_l] = f32_to_bf16(p);   // transpose through LDS
      }
    }
    rsum += __shfl_xor(rsum, 16, 32);
    l_run = l_run * alpha + rsum;
    m_run = m_new;

    // ---- reload P as WMMA A-frag (row = query = lane&15, K = 32 keys) ----
    FragB16 Pf;
    Pf.q[0] = *(const u32x4*)(myP + l16 * 32 +      8 * hs);
    Pf.q[1] = *(const u32x4*)(myP + l16 * 32 + 16 + 8 * hs);

    // ---- V B-frags: column = dim lane&15 (+16j), elems -> key in table order ----
    FragB16 v0, v1, v2, v3;
#pragma unroll
    for (int s = 0; s < 16; ++s) {
      const int kk = (s < 8) ? (8 * hs + s) : (16 + 8 * hs + (s - 8));
      int vrow = kb * 32 + kk;
      if (vrow > T_ - 1) vrow = T_ - 1;            // P==0 there
      const u16* vbase = base + (size_t)vrow * LD3C + 2 * C_ + l16;
      v0.h[s] = vbase[0]; v1.h[s] = vbase[16]; v2.h[s] = vbase[32]; v3.h[s] = vbase[48];
    }

    // ---- O = alpha*O + P @ V ----
#pragma unroll
    for (int r = 0; r < 8; ++r) {
      const float a_r = __shfl(alpha, r + 8 * hs, 32);
      O0[r] *= a_r; O1[r] *= a_r; O2[r] *= a_r; O3[r] *= a_r;
    }
    O0 = __builtin_amdgcn_wmma_f32_16x16x32_bf16(false, Pf.v, false, v0.v, (short)0, O0, false, false);
    O1 = __builtin_amdgcn_wmma_f32_16x16x32_bf16(false, Pf.v, false, v1.v, (short)0, O1, false, false);
    O2 = __builtin_amdgcn_wmma_f32_16x16x32_bf16(false, Pf.v, false, v2.v, (short)0, O2, false, false);
    O3 = __builtin_amdgcn_wmma_f32_16x16x32_bf16(false, Pf.v, false, v3.v, (short)0, O3, false, false);
  }

  // ---- normalize by 1/l and store y[b, q, h*64 + d] as bf16 ----
#pragma unroll
  for (int r = 0; r < 8; ++r) {
    const int m = r + 8 * hs;
    const float linv = 1.f / __shfl(l_run, m, 32);
    const size_t row = ((size_t)b * T_ + qi * 16 + m) * C_ + h * D_ + l16;
    y[row +  0] = f32_to_bf16(O0[r] * linv);
    y[row + 16] = f32_to_bf16(O1[r] * linv);
    y[row + 32] = f32_to_bf16(O2[r] * linv);
    y[row + 48] = f32_to_bf16(O3[r] * linv);
  }
}

// ---------------------------------------------------------------------------
// Launch
// ---------------------------------------------------------------------------
extern "C" void kernel_launch(void* const* d_in, const int* in_sizes, int n_in,
                              void* d_out, int out_size, void* d_ws, size_t ws_size,
                              hipStream_t stream) {
  const float* x  = (const float*)d_in[0];   // [B,T,C]
  const float* Wa = (const float*)d_in[1];   // [C,3C]
  const float* Wp = (const float*)d_in[2];   // [C,C]
  float* out = (float*)d_out;                // [B,T,C] fp32

  char* ws = (char*)d_ws;                    // ~50.3 MB used
  u16* xb   = (u16*)(ws + 0);                //  8 MB : x  bf16
  u16* wab  = (u16*)(ws + 8388608);          //  6 MB : W_attn bf16
  u16* wpb  = (u16*)(ws + 14680064);         //  2 MB : W_proj bf16
  u16* qkvb = (u16*)(ws + 16777216);         // 24 MB : qkv bf16 [4096,3072]
  u16* yb   = (u16*)(ws + 41943040);         //  8 MB : y bf16 [4096,1024]

  const int nX = B_ * T_ * C_;               // 4194304
  const int nA = C_ * 3 * C_;                // 3145728
  const int nP = C_ * C_;                    // 1048576

  cvt_f32_bf16_kernel<<<4096, 256, 0, stream>>>(x,  xb,  nX);
  cvt_f32_bf16_kernel<<<4096, 256, 0, stream>>>(Wa, wab, nA);
  cvt_f32_bf16_kernel<<<2048, 256, 0, stream>>>(Wp, wpb, nP);

  // qkv = x @ W_attn : M=4096, N=3072 -> 32 x 24 tiles of 128x128
  gemm_bf16_wmma<true><<<768, 256, 0, stream>>>(xb, wab, (void*)qkvb,
                                                B_ * T_, 3 * C_, C_);

  // flash attention: B*H*(T/16) = 4096 wave-jobs, 4 waves per block
  attn_flash_kernel<<<1024, 128, 0, stream>>>(qkvb, yb);

  // out = y @ W_proj : M=4096, N=1024 -> 32 x 8 tiles of 128x128
  gemm_bf16_wmma<false><<<256, 256, 0, stream>>>(yb, wpb, (void*)out,
                                                 B_ * T_, C_, C_);
}
